// Dropx_LayerScale_Block_7773890806046
// MI455X (gfx1250) — matrix operand order repaired and checked by reading, not compiled, not run on hardware
//
#include <hip/hip_runtime.h>
#include <cstdint>

typedef __attribute__((ext_vector_type(16))) __bf16 v16bf;
typedef __attribute__((ext_vector_type(8)))  float  v8f;

#define DEV __device__ __forceinline__

DEV unsigned short f2bf(float f){
  unsigned int x = __float_as_uint(f);
  unsigned int r = x + 0x7fffu + ((x >> 16) & 1u);
  return (unsigned short)(r >> 16);
}
DEV float bf2f(unsigned short h){
  return __uint_as_float(((unsigned int)h) << 16);
}

union FragBF { v16bf v; unsigned int u[8]; };

// ---------------- weight f32 -> bf16 convert ----------------
__global__ void __launch_bounds__(256) cvt_f32_bf16(const float* __restrict__ src,
                                                    unsigned short* __restrict__ dst, int n){
  int i = blockIdx.x * 1024 + threadIdx.x;
  #pragma unroll
  for (int j = 0; j < 4; ++j){
    int k = i + j * 256;
    if (k < n) dst[k] = f2bf(src[k]);
  }
}

// ---------------- gather + layernorm (GATHER) or plain layernorm ----------------
template<bool GATHER>
__global__ void __launch_bounds__(256) ln_kernel(
    const float* __restrict__ src, const int* __restrict__ keep,
    const float* __restrict__ g, const float* __restrict__ bta,
    float* __restrict__ xg, unsigned short* __restrict__ hb,
    int NP, int NF)
{
  __shared__ float red[256];
  const int tid = threadIdx.x;
  long row;
  if (GATHER){
    int b = blockIdx.x / NP, n = blockIdx.x - b * NP;
    int s = keep[b * NP + n];
    row = (long)b * NF + s;
  } else {
    row = blockIdx.x;
  }
  const float* p = src + row * 768;
  float v0 = p[tid], v1 = p[tid + 256], v2 = p[tid + 512];
  red[tid] = v0 + v1 + v2; __syncthreads();
  for (int st = 128; st > 0; st >>= 1){ if (tid < st) red[tid] += red[tid + st]; __syncthreads(); }
  float mean = red[0] * (1.0f / 768.0f);
  __syncthreads();
  float d0 = v0 - mean, d1 = v1 - mean, d2 = v2 - mean;
  red[tid] = d0*d0 + d1*d1 + d2*d2; __syncthreads();
  for (int st = 128; st > 0; st >>= 1){ if (tid < st) red[tid] += red[tid + st]; __syncthreads(); }
  float rstd = rsqrtf(red[0] * (1.0f / 768.0f) + 1e-5f);
  long o = (long)blockIdx.x * 768;
  if (GATHER){
    xg[o + tid] = v0; xg[o + tid + 256] = v1; xg[o + tid + 512] = v2;
  }
  hb[o + tid]       = f2bf(d0 * rstd * g[tid]       + bta[tid]);
  hb[o + tid + 256] = f2bf(d1 * rstd * g[tid + 256] + bta[tid + 256]);
  hb[o + tid + 512] = f2bf(d2 * rstd * g[tid + 512] + bta[tid + 512]);
}

// ---------------- generic bf16 WMMA GEMM, block 64x128, BK=32, 8 waves of 32x32 ----------------
// Double-buffered LDS pipeline: fetch next tile -> WMMA on current -> commit next -> barrier.
// EPI: 1=QKV split, 2=S store, 3=AV out, 4=proj+res, 5=fc1 gelu, 6=fc2+res
template<int EPI, bool BT>
__global__ void __launch_bounds__(256) gemm_bf16_k(
    const unsigned short* __restrict__ A, const unsigned short* __restrict__ Bm,
    int M, int N, int K, int lda, int ldb,
    long bsA, long bsB,
    void* __restrict__ out0, void* __restrict__ out1, void* __restrict__ out2,
    const float* __restrict__ bias, const float* __restrict__ gamma,
    const float* __restrict__ resid,
    float scaleq, int NP, int H, int MP, int KrealB, int NrealB)
{
  __shared__ unsigned short lds_a[2][64 * 40];
  __shared__ unsigned short lds_b[2][16 * 128 * 2];   // [kpair][col][k&1]
  const int tid  = threadIdx.x;
  const int lane = tid & 31, w = tid >> 5;
  const int wm = w >> 2, wn = w & 3;
  const int m0 = blockIdx.y * 64, n0 = blockIdx.x * 128;
  const int bz = blockIdx.z;
  const unsigned short* Ab = A + bsA * (long)bz;
  const unsigned short* Bb = Bm + bsB * (long)bz;

  v8f acc[2][2];
  #pragma unroll
  for (int s = 0; s < 2; ++s)
    #pragma unroll
    for (int t = 0; t < 2; ++t)
      acc[s][t] = (v8f){0.f,0.f,0.f,0.f,0.f,0.f,0.f,0.f};

  const int ar = tid >> 2, akc = (tid & 3) * 8;      // A loader: 64 rows x 4 chunks of 8
  const int bcol = tid >> 1, bdc = (tid & 1) * 16;   // BT loader: 128 cols x 2 chunks of 16
  const int bkr  = tid >> 3, bnc = (tid & 7) * 16;   // normal B loader: 32 rows x 8 chunks of 16
  const int hb8 = (lane >> 4) * 8;
  const int ml  = lane & 15;

  auto fetchA = [&](int k0, uint4& r){
    r = make_uint4(0u, 0u, 0u, 0u);
    int gr = m0 + ar;
    if (gr < M){
      const unsigned short* sa = Ab + (long)gr * lda + k0 + akc;
      r = *reinterpret_cast<const uint4*>(sa);
      if (k0 + 32 < K) __builtin_prefetch(sa + 32, 0, 1);
    }
  };
  auto fetchB = [&](int k0, uint4& r0, uint4& r1){
    r0 = make_uint4(0u, 0u, 0u, 0u);
    r1 = make_uint4(0u, 0u, 0u, 0u);
    if (BT){
      int gcol = n0 + bcol;
      if (gcol < NrealB){
        const uint4* sb = reinterpret_cast<const uint4*>(Bb + (long)gcol * ldb + k0 + bdc);
        r0 = sb[0]; r1 = sb[1];
      }
    } else {
      int gk = k0 + bkr, gc = n0 + bnc;
      if (gk < KrealB && gc < NrealB){
        const uint4* sb = reinterpret_cast<const uint4*>(Bb + (long)gk * ldb + gc);
        r0 = sb[0]; r1 = sb[1];
      }
    }
  };
  auto commit = [&](int buf, uint4 a, uint4 b0, uint4 b1){
    *reinterpret_cast<uint4*>(&lds_a[buf][ar * 40 + akc]) = a;
    unsigned short tmp[16];
    *reinterpret_cast<uint4*>(tmp)     = b0;
    *reinterpret_cast<uint4*>(tmp + 8) = b1;
    if (BT){
      #pragma unroll
      for (int i = 0; i < 16; ++i){
        int kk = bdc + i;
        lds_b[buf][((kk >> 1) * 128 + bcol) * 2 + (kk & 1)] = tmp[i];
      }
    } else {
      #pragma unroll
      for (int i = 0; i < 16; ++i){
        int cc = bnc + i;
        lds_b[buf][((bkr >> 1) * 128 + cc) * 2 + (bkr & 1)] = tmp[i];
      }
    }
  };

  // prologue: stage tile 0
  {
    uint4 a0, b0, b1;
    fetchA(0, a0); fetchB(0, b0, b1);
    commit(0, a0, b0, b1);
  }
  __syncthreads();

  int buf = 0;
  for (int k0 = 0; k0 < K; k0 += 32){
    const bool more = (k0 + 32 < K);
    uint4 na = make_uint4(0u,0u,0u,0u), nb0 = na, nb1 = na;
    if (more){ fetchA(k0 + 32, na); fetchB(k0 + 32, nb0, nb1); }

    // ---- fragments (contiguous k = 16*(lane/16) + s, same in A and B) ----
    FragBF af[2], bfr[2];
    #pragma unroll
    for (int s = 0; s < 2; ++s){
      const unsigned short* pa = &lds_a[buf][(wm * 32 + s * 16 + ml) * 40];
      #pragma unroll
      for (int v = 0; v < 8; ++v){
        int kk = hb8 * 2 + v * 2;
        af[s].u[v] = *reinterpret_cast<const unsigned int*>(pa + kk);
      }
    }
    #pragma unroll
    for (int t = 0; t < 2; ++t){
      int col = wn * 32 + t * 16 + ml;
      #pragma unroll
      for (int j = 0; j < 8; ++j)
        bfr[t].u[j] = *reinterpret_cast<const unsigned int*>(&lds_b[buf][((hb8 + j) * 128 + col) * 2]);
    }
    #pragma unroll
    for (int s = 0; s < 2; ++s)
      #pragma unroll
      for (int t = 0; t < 2; ++t)
        acc[s][t] = __builtin_amdgcn_wmma_f32_16x16x32_bf16(
            false, af[s].v, false, bfr[t].v, (short)0, acc[s][t], false, false);

    if (more) commit(buf ^ 1, na, nb0, nb1);
    __syncthreads();
    buf ^= 1;
  }

  // ---- epilogue ----
  const int halfl = lane >> 4;
  #pragma unroll
  for (int s = 0; s < 2; ++s)
  for (int t = 0; t < 2; ++t)
  #pragma unroll
  for (int r = 0; r < 8; ++r){
    int row = m0 + wm * 32 + s * 16 + halfl * 8 + r;
    int col = n0 + wn * 32 + t * 16 + ml;
    if (row >= M || col >= N) continue;
    float val = acc[s ^ ((buf & 0) ? 0 : 0)][t][r]; // acc[s][t][r]
    val = acc[s][t][r];
    if constexpr (EPI == 1){
      int which = col / 768, c = col - which * 768;
      int h = c >> 6, d = c & 63;
      int b = row / NP, n = row - b * NP;
      long idx = ((long)(b * H + h) * NP + n) * 64 + d;
      unsigned short* dst = (which == 0) ? (unsigned short*)out0
                          : (which == 1) ? (unsigned short*)out1
                                         : (unsigned short*)out2;
      dst[idx] = f2bf(which == 0 ? val * scaleq : val);
    } else if constexpr (EPI == 2){
      ((unsigned short*)out0)[((long)bz * M + row) * N + col] = f2bf(val);
    } else if constexpr (EPI == 3){
      int b = bz / H, h = bz - b * H;
      ((unsigned short*)out0)[((long)(b * NP + row)) * 768 + h * 64 + col] = f2bf(val);
    } else if constexpr (EPI == 4 || EPI == 6){
      long i = (long)row * N + col;
      ((float*)out0)[i] = resid[i] + gamma[col] * (val + bias[col]);
    } else if constexpr (EPI == 5){
      long i = (long)row * N + col;
      float a = val + bias[col];
      ((unsigned short*)out0)[i] = f2bf(0.5f * a * (1.0f + erff(a * 0.70710678118654752f)));
    }
  }
}

// ---------------- talking-heads: mix(W_l)+softmax+mix(W_w), per (b,n) token ----------------
__global__ void __launch_bounds__(256) talking_softmax_k(
    const unsigned short* __restrict__ S, unsigned short* __restrict__ A2,
    const float* __restrict__ Wl, const float* __restrict__ bl,
    const float* __restrict__ Ww, const float* __restrict__ bw,
    int NP, int MP)
{
  __shared__ float sS[12][521];
  __shared__ float sP[12][521];
  __shared__ float red[256];
  __shared__ float sWl[144], sWw[144], sbl[12], sbw[12];
  __shared__ float inv12[12];
  const int tid = threadIdx.x;
  const int b = blockIdx.x / NP, n = blockIdx.x - b * NP;
  if (tid < 144){ sWl[tid] = Wl[tid]; sWw[tid] = Ww[tid]; }
  if (tid < 12){ sbl[tid] = bl[tid]; sbw[tid] = bw[tid]; }
  for (int idx = tid; idx < 12 * NP; idx += 256){
    int h = idx / NP, m = idx - h * NP;
    sS[h][m] = bf2f(S[((long)(b * 12 + h) * NP + n) * MP + m]);
  }
  __syncthreads();
  for (int m = tid; m < NP; m += 256){
    float v[12];
    #pragma unroll
    for (int h = 0; h < 12; ++h) v[h] = sS[h][m];
    #pragma unroll
    for (int g = 0; g < 12; ++g){
      float a = sbl[g];
      #pragma unroll
      for (int h = 0; h < 12; ++h) a += v[h] * sWl[h * 12 + g];
      sP[g][m] = a;
    }
  }
  __syncthreads();
  for (int g = 0; g < 12; ++g){
    float lm = -3.0e38f;
    for (int m = tid; m < NP; m += 256) lm = fmaxf(lm, sP[g][m]);
    red[tid] = lm; __syncthreads();
    for (int st = 128; st > 0; st >>= 1){ if (tid < st) red[tid] = fmaxf(red[tid], red[tid + st]); __syncthreads(); }
    float gm = red[0]; __syncthreads();
    float ls = 0.f;
    for (int m = tid; m < NP; m += 256){ float e = expf(sP[g][m] - gm); sP[g][m] = e; ls += e; }
    red[tid] = ls; __syncthreads();
    for (int st = 128; st > 0; st >>= 1){ if (tid < st) red[tid] += red[tid + st]; __syncthreads(); }
    if (tid == 0) inv12[g] = 1.0f / red[0];
    __syncthreads();
  }
  for (int m = tid; m < MP; m += 256){
    float o[12];
    if (m < NP){
      float p[12];
      #pragma unroll
      for (int g = 0; g < 12; ++g) p[g] = sP[g][m] * inv12[g];
      #pragma unroll
      for (int g2 = 0; g2 < 12; ++g2){
        float a = sbw[g2];
        #pragma unroll
        for (int g = 0; g < 12; ++g) a += p[g] * sWw[g * 12 + g2];
        o[g2] = a;
      }
    } else {
      #pragma unroll
      for (int g2 = 0; g2 < 12; ++g2) o[g2] = 0.f;
    }
    #pragma unroll
    for (int g2 = 0; g2 < 12; ++g2)
      A2[((long)(b * 12 + g2) * NP + n) * MP + m] = f2bf(o[g2]);
  }
}

extern "C" void kernel_launch(void* const* d_in, const int* in_sizes, int n_in,
                              void* d_out, int out_size, void* d_ws, size_t ws_size,
                              hipStream_t stream)
{
  (void)in_sizes; (void)n_in; (void)out_size; (void)ws_size;
  const int Bb = 16, NF = 577, H = 12, NP = 520, MP = 544;
  const long T = (long)Bb * NP;  // 8320

  const float* x      = (const float*)d_in[0];
  const int*   keep   = (const int*)  d_in[1];
  const float* Wqkv   = (const float*)d_in[2];
  const float* Wproj  = (const float*)d_in[3];
  const float* bproj  = (const float*)d_in[4];
  const float* Wl     = (const float*)d_in[5];
  const float* bl     = (const float*)d_in[6];
  const float* Ww     = (const float*)d_in[7];
  const float* bw     = (const float*)d_in[8];
  const float* ln1g   = (const float*)d_in[9];
  const float* ln1b   = (const float*)d_in[10];
  const float* ln2g   = (const float*)d_in[11];
  const float* ln2b   = (const float*)d_in[12];
  const float* gamma1 = (const float*)d_in[13];
  const float* gamma2 = (const float*)d_in[14];
  const float* Wfc1   = (const float*)d_in[15];
  const float* bfc1   = (const float*)d_in[16];
  const float* Wfc2   = (const float*)d_in[17];
  const float* bfc2   = (const float*)d_in[18];

  char* ws = (char*)d_ws;
  size_t off = 0;
  auto alloc = [&](size_t bytes) -> char* {
    char* p = ws + off;
    off = (off + bytes + 255) & ~(size_t)255;
    return p;
  };
  float*          xg     = (float*)         alloc((size_t)T * 768 * 4);
  unsigned short* hbuf   = (unsigned short*)alloc((size_t)T * 768 * 2);
  unsigned short* wqkvb  = (unsigned short*)alloc((size_t)768 * 2304 * 2);
  unsigned short* wprojb = (unsigned short*)alloc((size_t)768 * 768 * 2);
  unsigned short* wfc1b  = (unsigned short*)alloc((size_t)768 * 3072 * 2);
  unsigned short* wfc2b  = (unsigned short*)alloc((size_t)3072 * 768 * 2);
  unsigned short* qb     = (unsigned short*)alloc((size_t)Bb * H * NP * 64 * 2);
  unsigned short* kb     = (unsigned short*)alloc((size_t)Bb * H * NP * 64 * 2);
  unsigned short* vb     = (unsigned short*)alloc((size_t)Bb * H * NP * 64 * 2);
  unsigned short* Sbuf   = (unsigned short*)alloc((size_t)Bb * H * NP * MP * 2);
  unsigned short* A2buf  = (unsigned short*)alloc((size_t)Bb * H * NP * MP * 2);
  float*          x1     = (float*)         alloc((size_t)T * 768 * 4);
  unsigned short* attn_out = qb;   // alias: q dead after S GEMM
  unsigned short* h2b      = kb;   // alias: k dead after S GEMM
  unsigned short* hfc      = Sbuf; // alias: S dead after softmax

  auto cvt = [&](const float* s, unsigned short* d, int n){
    cvt_f32_bf16<<<(n + 1023) / 1024, 256, 0, stream>>>(s, d, n);
  };
  cvt(Wqkv,  wqkvb,  768 * 2304);
  cvt(Wproj, wprojb, 768 * 768);
  cvt(Wfc1,  wfc1b,  768 * 3072);
  cvt(Wfc2,  wfc2b,  3072 * 768);

  ln_kernel<true><<<(int)T, 256, 0, stream>>>(x, keep, ln1g, ln1b, xg, hbuf, NP, NF);

  // QKV: [8320,768] @ [768,2304]
  gemm_bf16_k<1, false><<<dim3(18, 130, 1), 256, 0, stream>>>(
      hbuf, wqkvb, 8320, 2304, 768, 768, 2304, 0, 0,
      qb, kb, vb, nullptr, nullptr, nullptr, 0.125f, NP, H, MP, 768, 2304);

  // S = q @ k^T per (b,h): [520,64] @ [64,544(pad)]
  gemm_bf16_k<2, true><<<dim3(5, 9, 192), 256, 0, stream>>>(
      qb, kb, NP, MP, 64, 64, 64, (long)NP * 64, (long)NP * 64,
      Sbuf, nullptr, nullptr, nullptr, nullptr, nullptr, 0.f, NP, H, MP, 64, NP);

  talking_softmax_k<<<(int)T, 256, 0, stream>>>(Sbuf, A2buf, Wl, bl, Ww, bw, NP, MP);

  // out = A2 @ v per (b,h): [520,544] @ [544(pad),64]
  gemm_bf16_k<3, false><<<dim3(1, 9, 192), 256, 0, stream>>>(
      A2buf, vb, NP, 64, MP, MP, 64, (long)NP * MP, (long)NP * 64,
      attn_out, nullptr, nullptr, nullptr, nullptr, nullptr, 0.f, NP, H, MP, NP, 64);

  // proj + layerscale residual
  gemm_bf16_k<4, false><<<dim3(6, 130, 1), 256, 0, stream>>>(
      attn_out, wprojb, 8320, 768, 768, 768, 768, 0, 0,
      x1, nullptr, nullptr, bproj, gamma1, xg, 0.f, NP, H, MP, 768, 768);

  ln_kernel<false><<<(int)T, 256, 0, stream>>>(x1, nullptr, ln2g, ln2b, nullptr, h2b, NP, NF);

  // fc1 + exact GELU
  gemm_bf16_k<5, false><<<dim3(24, 130, 1), 256, 0, stream>>>(
      h2b, wfc1b, 8320, 3072, 768, 768, 3072, 0, 0,
      hfc, nullptr, nullptr, bfc1, nullptr, nullptr, 0.f, NP, H, MP, 768, 3072);

  // fc2 + layerscale residual -> d_out (f32)
  gemm_bf16_k<6, false><<<dim3(6, 130, 1), 256, 0, stream>>>(
      hfc, wfc2b, 8320, 768, 3072, 3072, 768, 0, 0,
      d_out, nullptr, nullptr, bfc2, gamma2, x1, 0.f, NP, H, MP, 3072, 768);
}
